// DoraLinearLayer_41712722379292
// MI455X (gfx1250) — matrix operand order repaired and checked by reading, not compiled, not run on hardware
//
#include <hip/hip_runtime.h>

typedef __attribute__((ext_vector_type(16))) __bf16 v16bf;
typedef __attribute__((ext_vector_type(8)))  float  v8f;
typedef __attribute__((ext_vector_type(4)))  int    v4i;
typedef unsigned short u16;
typedef unsigned int   u32;

#define BM 128
#define BN 128
#define BK 32
#define KP 40   // halves per LDS row: 16B-aligned tile stores, 4B-aligned frag loads

#if __has_builtin(__builtin_amdgcn_global_load_async_to_lds_b128)
#define HAVE_ASYNC_BUILTIN 1
#else
#define HAVE_ASYNC_BUILTIN 0
#endif

typedef __attribute__((address_space(1))) v4i* gv4i_p;
typedef __attribute__((address_space(3))) v4i* lv4i_p;

// 16-byte async global->LDS copy (ASYNCcnt-tracked, no VGPR staging)
__device__ __forceinline__ void async_cp16(const u16* __restrict__ g, u16* l) {
#if HAVE_ASYNC_BUILTIN
  __builtin_amdgcn_global_load_async_to_lds_b128((gv4i_p)g, (lv4i_p)l, 0, 0);
#else
  // VGLOBAL async encoding: VDST = LDS byte address (addr[31:0] of generic
  // shared pointer per the flat->LDS aperture rule), VADDR = 64-bit global.
  asm volatile("global_load_async_to_lds_b128 %0, %1, off"
               :: "v"((u32)(uintptr_t)l), "v"(g)
               : "memory");
#endif
}

__device__ __forceinline__ void wait_async0() {
#if __has_builtin(__builtin_amdgcn_s_wait_asynccnt)
  __builtin_amdgcn_s_wait_asynccnt(0);
#else
  asm volatile("s_wait_asynccnt 0x0" ::: "memory");
#endif
}

__device__ __forceinline__ u16 f2bf(float f) {
  u32 u = __float_as_uint(f);
  u = (u + 0x7FFFu + ((u >> 16) & 1u)) >> 16;   // round-to-nearest-even
  return (u16)u;
}

// ---------------- fp32 -> bf16 bulk convert (8 elems/thread) ----------------
__global__ __launch_bounds__(256) void cvt_bf16(const float* __restrict__ src,
                                                u16* __restrict__ dst,
                                                long long n) {
  long long i = ((long long)blockIdx.x * blockDim.x + threadIdx.x) * 8;
  if (i >= n) return;
  float4 a = *(const float4*)&src[i];
  float4 b = *(const float4*)&src[i + 4];
  union { u16 s[8]; uint4 u; } p;
  p.s[0] = f2bf(a.x); p.s[1] = f2bf(a.y); p.s[2] = f2bf(a.z); p.s[3] = f2bf(a.w);
  p.s[4] = f2bf(b.x); p.s[5] = f2bf(b.y); p.s[6] = f2bf(b.z); p.s[7] = f2bf(b.w);
  *(uint4*)&dst[i] = p.u;
}

// ------- scale[o] = magnitude[o] / || base_w[o,:] + 2*(B@A)[o,:] ||  (1 wave/row) -------
__global__ __launch_bounds__(256) void dora_scale(const float* __restrict__ W,
                                                  const float* __restrict__ Aw,
                                                  const float* __restrict__ Bw,
                                                  const float* __restrict__ mag,
                                                  float* __restrict__ Sc,
                                                  int N, int K) {
  int wave = threadIdx.x >> 5, lane = threadIdx.x & 31;
  int o = blockIdx.x * 8 + wave;
  if (o >= N) return;
  float br[16];
#pragma unroll
  for (int r = 0; r < 16; ++r) br[r] = Bw[(size_t)o * 16 + r];
  float acc = 0.f;
  for (int i = lane; i < K; i += 32) {
    float lw = 0.f;
#pragma unroll
    for (int r = 0; r < 16; ++r) lw += br[r] * Aw[(size_t)r * K + i];
    float w = W[(size_t)o * K + i] + 2.0f * lw;
    acc += w * w;
  }
#pragma unroll
  for (int off = 16; off; off >>= 1) acc += __shfl_xor(acc, off, 32);
  if (lane == 0) Sc[o] = mag[o] / sqrtf(acc);
}

// ---------------- u = x @ A^T  -> [T,16]  (1 wave per token) ----------------
__global__ __launch_bounds__(256) void dora_xa(const float* __restrict__ X,
                                               const float* __restrict__ Aw,
                                               float* __restrict__ U,
                                               int T, int K) {
  int wave = threadIdx.x >> 5, lane = threadIdx.x & 31;
  int t = blockIdx.x * 8 + wave;
  if (t >= T) return;
  float acc[16];
#pragma unroll
  for (int r = 0; r < 16; ++r) acc[r] = 0.f;
  for (int i = lane; i < K; i += 32) {
    float xv = X[(size_t)t * K + i];
#pragma unroll
    for (int r = 0; r < 16; ++r) acc[r] += xv * Aw[(size_t)r * K + i];
  }
#pragma unroll
  for (int r = 0; r < 16; ++r) {
#pragma unroll
    for (int off = 16; off; off >>= 1) acc[r] += __shfl_xor(acc[r], off, 32);
  }
  if (lane == 0) {
#pragma unroll
    for (int r = 0; r < 16; ++r) U[(size_t)t * 16 + r] = acc[r];
  }
}

// ---------------- main fused GEMM + DoRA epilogue ----------------
// out[t,o] = (s[o]-1) * <x[t,:], base_w[o,:]> + 2*s[o] * <u[t,:], B[o,:]>
__global__ __launch_bounds__(256) void dora_gemm(const u16* __restrict__ Xb,
                                                 const u16* __restrict__ Wb,
                                                 const float* __restrict__ U,
                                                 const float* __restrict__ LB,
                                                 const float* __restrict__ Sc,
                                                 float* __restrict__ Out,
                                                 int T, int N, int K) {
  __shared__ u16 As[2][BM * KP];    // 2 x 10 KB
  __shared__ u16 Bs[2][BN * KP];    // 2 x 10 KB
  __shared__ float Us[BM * 16];     // 8 KB

  const int tid  = threadIdx.x;
  const int lane = tid & 31;
  const int wave = tid >> 5;
  const int wm = wave >> 1, wn = wave & 1;   // 4x2 wave grid
  const int m0 = wm * 32, n0 = wn * 64;      // wave tile: 32x64
  const int mBase = blockIdx.y * BM;
  const int nBase = blockIdx.x * BN;
  const int half = lane >> 4;
  const int l16  = lane & 15;

  const int row4 = tid >> 2;   // 0..63
  const int kq   = tid & 3;    // 8-half chunk within 32-wide K tile

  // stage this block's u-slice (128 rows x 16) into LDS
  {
    int row = tid >> 1, c = (tid & 1) * 8;
    float4 a = *(const float4*)&U[(size_t)(mBase + row) * 16 + c];
    float4 b = *(const float4*)&U[(size_t)(mBase + row) * 16 + c + 4];
    *(float4*)&Us[row * 16 + c]     = a;
    *(float4*)&Us[row * 16 + c + 4] = b;
  }

  v8f acc[2][4];
#pragma unroll
  for (int i = 0; i < 2; ++i)
#pragma unroll
    for (int j = 0; j < 4; ++j)
      acc[i][j] = (v8f){0.f, 0.f, 0.f, 0.f, 0.f, 0.f, 0.f, 0.f};

  union AV { u32 u[8]; v16bf v; };

  // ---- async tile copy: 4 x b128 per thread per K-step ----
  auto load_tiles = [&](int kt, int buf) {
#pragma unroll
    for (int h = 0; h < 2; ++h) {
      int row = row4 + h * 64;
      async_cp16(&Xb[(size_t)(mBase + row) * K + kt + kq * 8],
                 &As[buf][row * KP + kq * 8]);
      async_cp16(&Wb[(size_t)(nBase + row) * K + kt + kq * 8],
                 &Bs[buf][row * KP + kq * 8]);
    }
  };

  // prologue: fill buffer 0
  load_tiles(0, 0);
  wait_async0();
  __syncthreads();

  int buf = 0;
  for (int kt = 0; kt < K; kt += BK) {
    const bool more = (kt + BK) < K;
    if (more) load_tiles(kt + BK, buf ^ 1);   // overlap next tile with compute

    // A fragments: 16-bit A 16x32 layout (contiguous K order, matching B):
    //   VGPR v holds K = 2v,2v+1 (+16 for lanes 16-31) ; M = lane%16
    AV af[2];
#pragma unroll
    for (int mt = 0; mt < 2; ++mt) {
      int m = m0 + mt * 16 + l16;
#pragma unroll
      for (int v = 0; v < 8; ++v) {
        int kb = 2 * v + 16 * half;
        af[mt].u[v] = *(const u32*)&As[buf][m * KP + kb];
      }
    }
    // B fragments: 16-bit B 32x16: VGPR v holds K = 2v,2v+1 (+16 for lanes 16-31); N = lane%16
    AV bf[4];
#pragma unroll
    for (int nt = 0; nt < 4; ++nt) {
      int n = n0 + nt * 16 + l16;
#pragma unroll
      for (int v = 0; v < 8; ++v) {
        int kb = 2 * v + 16 * half;
        bf[nt].u[v] = *(const u32*)&Bs[buf][n * KP + kb];
      }
    }

#pragma unroll
    for (int mt = 0; mt < 2; ++mt)
#pragma unroll
      for (int nt = 0; nt < 4; ++nt)
        acc[mt][nt] = __builtin_amdgcn_wmma_f32_16x16x32_bf16(
            false, af[mt].v, false, bf[nt].v, (short)0, acc[mt][nt], false, false);

    if (more) wait_async0();   // our next-buffer copies have landed in LDS
    __syncthreads();           // everyone done reading buf + everyone's copies visible
    buf ^= 1;
  }

  // epilogue: C/D layout — VGPR j holds M = j + 8*(lane/16), N = lane%16
#pragma unroll
  for (int nt = 0; nt < 4; ++nt) {
    int o = nBase + n0 + nt * 16 + l16;
    float s = Sc[o];
    float lb[16];
#pragma unroll
    for (int q = 0; q < 4; ++q) {
      float4 t4 = *(const float4*)&LB[(size_t)o * 16 + q * 4];
      lb[q * 4 + 0] = t4.x; lb[q * 4 + 1] = t4.y;
      lb[q * 4 + 2] = t4.z; lb[q * 4 + 3] = t4.w;
    }
#pragma unroll
    for (int mt = 0; mt < 2; ++mt) {
#pragma unroll
      for (int j = 0; j < 8; ++j) {
        int ml = m0 + mt * 16 + 8 * half + j;
        float lora = 0.f;
#pragma unroll
        for (int r = 0; r < 16; ++r) lora += Us[ml * 16 + r] * lb[r];
        float b = acc[mt][nt][j];
        Out[(size_t)(mBase + ml) * N + o] = (s - 1.0f) * b + 2.0f * s * lora;
      }
    }
  }
}

extern "C" void kernel_launch(void* const* d_in, const int* in_sizes, int n_in,
                              void* d_out, int out_size, void* d_ws, size_t ws_size,
                              hipStream_t stream) {
  const float* x   = (const float*)d_in[0];   // [T, K]
  const float* Aw  = (const float*)d_in[1];   // [r, K]
  const float* Bw  = (const float*)d_in[2];   // [N, r]
  const float* W   = (const float*)d_in[3];   // [N, K]
  const float* mag = (const float*)d_in[4];   // [N]
  float* out = (float*)d_out;

  const int N = in_sizes[4];                  // 4096
  const int r = in_sizes[2] / N;              // 16
  const int K = in_sizes[1] / r;              // 4096
  const int T = in_sizes[0] / K;              // 8192

  char* ws = (char*)d_ws;
  float* U  = (float*)ws;                                  // T*16*4   = 512 KB
  float* Sc = (float*)(ws + (size_t)T * 16 * 4);           // N*4      = 16 KB
  u16*  Xb  = (u16*)(ws + (size_t)T * 16 * 4 + (size_t)N * 4);
  u16*  Wb  = Xb + (size_t)T * K;                          // bf16 copies: 64+32 MB (L2-resident)

  long long nx = (long long)T * K, nw = (long long)N * K;
  cvt_bf16<<<(unsigned)((nx / 8 + 255) / 256), 256, 0, stream>>>(x, Xb, nx);
  cvt_bf16<<<(unsigned)((nw / 8 + 255) / 256), 256, 0, stream>>>(W, Wb, nw);
  dora_scale<<<(N + 7) / 8, 256, 0, stream>>>(W, Aw, Bw, mag, Sc, N, K);
  dora_xa<<<(T + 7) / 8, 256, 0, stream>>>(x, Aw, U, T, K);

  dim3 grid(N / BN, T / BM);   // (32, 64)
  dora_gemm<<<grid, 256, 0, stream>>>(Xb, Wb, U, Bw, Sc, out, T, N, K);
}
